// ContactNet_56985626083683
// MI455X (gfx1250) — matrix operand order repaired and checked by reading, not compiled, not run on hardware
//
#include <hip/hip_runtime.h>
#include <hip/hip_bf16.h>

// ---------------------------------------------------------------------------
// ContactNet forward, MI455X (gfx1250, wave32, WMMA f32<-f16 16x16x32).
//
// Algebraic restructuring (see round 0): the 278-GFLOP wf layer collapses to
//   out[b,o,f] = (f<1024 ? g[b,f]*wsum[o] : P[b,o,f-1024]) + c2[b,o]
// and h2 is generated on the fly inside the w1 GEMM.
//
// Codegen restructuring (this round): all weights pre-packed to zero-padded
// f16 (A tiles = 2x b128 loads, no guards); activations stored channel-last
// with rows padded to 832 (B tiles = 2x b128 loads, epilogue = 1x b128 store,
// no guards); w1 K-loop split at the k=1024 tile boundary so tile generation
// is branchless and uniform.
// ---------------------------------------------------------------------------

typedef __attribute__((ext_vector_type(16))) _Float16 v16h;
typedef __attribute__((ext_vector_type(8)))  _Float16 v8h;
typedef __attribute__((ext_vector_type(8)))  float    v8f;
typedef __attribute__((ext_vector_type(4)))  float    v4f;

namespace cn {
constexpr int B   = 128;
constexpr int N   = 778;
constexpr int NPP = 832;   // point rows padded to 13*64 (grid covers all rows)
constexpr int DPE = 504;
constexpr int DPEP= 512;   // padded
constexpr int C1  = 64;
constexpr int C2  = 128;
constexpr int C3  = 1024;
constexpr int F   = 1088;  // C3 + C1
constexpr int O1  = 512;
constexpr int O2  = 256;
constexpr int O3  = 128;
constexpr int WF_LD  = N + DPE; // 1282 (source)
constexpr int WFA_R  = 784;     // packed wf rows for P-GEMM A (49*16)
constexpr int WFA_LD = 800;     // packed wf cols (25*32)
constexpr int NT64   = NPP / 64; // 13
}

// ---------------- device helpers ----------------

__device__ __forceinline__ v8f vzero8() {
  v8f z;
#pragma unroll
  for (int i = 0; i < 8; ++i) z[i] = 0.0f;
  return z;
}

__device__ __forceinline__ v8f wmma32(v16h a, v16h b, v8f c) {
  return __builtin_amdgcn_wmma_f32_16x16x32_f16(false, a, false, b, (short)0, c,
                                                false, false);
}

__device__ __forceinline__ unsigned f2ord(float f) {
  unsigned u = __float_as_uint(f);
  return (u & 0x80000000u) ? ~u : (u | 0x80000000u);
}
__device__ __forceinline__ float ord2f(unsigned u) {
  return __uint_as_float((u & 0x80000000u) ? (u & 0x7FFFFFFFu) : ~u);
}

// A tile 16x32, f16, K-contiguous rows. p = row base (+k0). Map:
//   halves h : K = 16g + h
__device__ __forceinline__ v16h loadA16(const _Float16* p, int g) {
  v8h a0 = *(const v8h*)(p + 16 * g);
  v8h a1 = *(const v8h*)(p + 16 * g + 8);
  v16h out;
#pragma unroll
  for (int i = 0; i < 8; ++i) { out[i] = a0[i]; out[8 + i] = a1[i]; }
  return out;
}

// B tile 32x16, f16, K-contiguous per column. p = col base (+k0).
//   halves h : K = 16g + h
__device__ __forceinline__ v16h loadBc(const _Float16* p, int g) {
  v8h b0 = *(const v8h*)(p + 16 * g);
  v8h b1 = *(const v8h*)(p + 16 * g + 8);
  v16h out;
#pragma unroll
  for (int i = 0; i < 8; ++i) { out[i] = b0[i]; out[8 + i] = b1[i]; }
  return out;
}

// B tile 32x16, f16, K strided by `ld` (lane-coalesced along n).
__device__ __forceinline__ v16h loadBs(const _Float16* p, int ld, int g) {
  v16h out;
#pragma unroll
  for (int h = 0; h < 16; ++h) out[h] = p[(size_t)(16 * g + h) * ld];
  return out;
}

// convert 8 f32 accum rows to f16 (biasv unused; no bias/bn/relu) and store as one b128
__device__ __forceinline__ void store_bn_relu(float* biasv, const v8f& acc,
                                              _Float16* dst) {
  v8h hv;
#pragma unroll
  for (int i = 0; i < 8; ++i) hv[i] = (_Float16)acc[i];
  *(v8h*)dst = hv;
}

// ---------------- pack kernels (run once per call) ----------------

__global__ void k_pack_f16(_Float16* __restrict__ dst, const float* __restrict__ src,
                           int srcRows, int srcCols, int srcLd, int dstCols,
                           int total) {
  int i = blockIdx.x * 256 + threadIdx.x;
  if (i >= total) return;
  int r = i / dstCols, c = i % dstCols;
  float v = (r < srcRows && c < srcCols) ? src[(size_t)r * srcLd + c] : 0.0f;
  dst[i] = (_Float16)v;
}

__global__ void k_pack_f32(float* __restrict__ dst, const float* __restrict__ src,
                           int srcN, int total) {
  int i = blockIdx.x * 256 + threadIdx.x;
  if (i >= total) return;
  dst[i] = (i < srcN) ? src[i] : 0.0f;
}

// ---------------- pointnet layer 1 ----------------
// pf[b][n][64] f16, rows >= 778 zeroed
__global__ void k_l1(const float* __restrict__ verts, const float* __restrict__ normals,
                     const float* __restrict__ pw1, const float* __restrict__ pb1,
                     const float* __restrict__ ps1, const float* __restrict__ pt1,
                     _Float16* __restrict__ pf) {
  using namespace cn;
  __shared__ float sw[C1 * 6], sb[C1], ss[C1], st[C1];
  int tid = threadIdx.x;
  for (int i = tid; i < C1 * 6; i += 256) sw[i] = pw1[i];
  for (int i = tid; i < C1; i += 256) { sb[i] = pb1[i]; ss[i] = ps1[i]; st[i] = pt1[i]; }
  __syncthreads();
  int idx = blockIdx.x * 256 + tid;
  if (idx >= B * NPP) return;
  int b = idx / NPP, n = idx % NPP;
  float x6[6] = {0, 0, 0, 0, 0, 0};
  bool valid = n < N;
  if (valid) {
    const float* v  = verts   + ((size_t)b * N + n) * 3;
    const float* nr = normals + ((size_t)b * N + n) * 3;
    x6[0] = v[0];  x6[1] = v[1];  x6[2] = v[2];
    x6[3] = nr[0]; x6[4] = nr[1]; x6[5] = nr[2];
  }
  _Float16* row = pf + (size_t)idx * C1;
#pragma unroll 2
  for (int j0 = 0; j0 < C1; j0 += 8) {
    v8h hv;
#pragma unroll
    for (int u = 0; u < 8; ++u) {
      int j = j0 + u;
      float s = 0.0f;
      if (valid) {
        s = sb[j];
#pragma unroll
        for (int c = 0; c < 6; ++c) s += sw[j * 6 + c] * x6[c];
        s = fmaxf(s * ss[j] + st[j], 0.0f);
      }
      hv[u] = (_Float16)s;
    }
    *(v8h*)(row + j0) = hv;
  }
}

// ---------------- generic per-batch GEMM, channel-last activations ----------
// Y[b][n][M] = relu(bn(Wh[MxK] @ X[b][n][K]^T)), grid (13, M/16, B), block 32
__global__ void k_gemm_bn_relu(const _Float16* __restrict__ Wh, int M, int K,
                               const _Float16* __restrict__ X,
                               const float* __restrict__ bias,
                               const float* __restrict__ scale,
                               const float* __restrict__ shift,
                               _Float16* __restrict__ Y) {
  using namespace cn;
  int lane = threadIdx.x;
  int g = lane >> 4, nl = lane & 15;
  int b = blockIdx.z;
  int m0 = blockIdx.y * 16;
  int n0b = blockIdx.x * 64;
  const _Float16* Arow = Wh + (size_t)(m0 + nl) * K;
  const _Float16* Xb = X + ((size_t)b * NPP + n0b + nl) * K;
  v8f acc[4];
#pragma unroll
  for (int j = 0; j < 4; ++j) acc[j] = vzero8();
  for (int k0 = 0; k0 < K; k0 += 32) {
    v16h a = loadA16(Arow + k0, g);
#pragma unroll
    for (int j = 0; j < 4; ++j) {
      v16h bt = loadBc(Xb + (size_t)(16 * j) * K + k0, g);
      acc[j] = wmma32(a, bt, acc[j]);
    }
  }
  int mb = m0 + 8 * g;
  float bi[8], sc[8], sh[8];
#pragma unroll
  for (int i = 0; i < 8; ++i) { bi[i] = bias[mb + i]; sc[i] = scale[mb + i]; sh[i] = shift[mb + i]; }
#pragma unroll
  for (int j = 0; j < 4; ++j) {
    int n = n0b + 16 * j + nl;
    v8h hv;
#pragma unroll
    for (int i = 0; i < 8; ++i) {
      float x = fmaxf((acc[j][i] + bi[i]) * sc[i] + sh[i], 0.0f);
      hv[i] = (_Float16)x;
    }
    *(v8h*)(Y + ((size_t)b * NPP + n) * M + mb) = hv;
  }
}

// ---------------- pointnet layer 3 + max pool (no store of h3) --------------
__global__ void k_gemm_bn_max(const _Float16* __restrict__ Wh,
                              const _Float16* __restrict__ X,
                              const float* __restrict__ bias,
                              const float* __restrict__ scale,
                              const float* __restrict__ shift,
                              unsigned* __restrict__ gmax) {
  using namespace cn;
  constexpr int M = C3, K = C2;
  int lane = threadIdx.x;
  int g = lane >> 4, nl = lane & 15;
  int b = blockIdx.z;
  int m0 = blockIdx.y * 16;
  int n0b = blockIdx.x * 64;
  const _Float16* Arow = Wh + (size_t)(m0 + nl) * K;
  const _Float16* Xb = X + ((size_t)b * NPP + n0b + nl) * K;
  v8f acc[4];
#pragma unroll
  for (int j = 0; j < 4; ++j) acc[j] = vzero8();
#pragma unroll
  for (int k0 = 0; k0 < K; k0 += 32) {
    v16h a = loadA16(Arow + k0, g);
#pragma unroll
    for (int j = 0; j < 4; ++j) {
      v16h bt = loadBc(Xb + (size_t)(16 * j) * K + k0, g);
      acc[j] = wmma32(a, bt, acc[j]);
    }
  }
#pragma unroll
  for (int v = 0; v < 8; ++v) {
    float mx = -3.4e38f;
#pragma unroll
    for (int j = 0; j < 4; ++j) {
      int n = n0b + 16 * j + nl;
      if (n < N) mx = fmaxf(mx, acc[j][v]);  // arithmetic guard only, no loads
    }
#pragma unroll
    for (int off = 1; off < 16; off <<= 1)
      mx = fmaxf(mx, __shfl_xor(mx, off, 32));
    int m = m0 + v + 8 * g;
    float y = (mx + bias[m]) * scale[m] + shift[m]; // bn scale > 0 -> monotone
    if (nl == 0) atomicMax(&gmax[(size_t)b * C3 + m], f2ord(y));
  }
}

__global__ void k_init_gmax(unsigned* __restrict__ gmax) {
  int i = blockIdx.x * 256 + threadIdx.x;
  if (i < cn::B * cn::C3) gmax[i] = 0x007FFFFFu; // f2ord(-inf)
}
__global__ void k_decode_gmax(const unsigned* __restrict__ gmax, float* __restrict__ gd) {
  int i = blockIdx.x * 256 + threadIdx.x;
  if (i < cn::B * cn::C3) gd[i] = ord2f(gmax[i]);
}

// wsumP[o] = sum_p wf[o,p] (o<778), padded zeros to 832
__global__ void k_wsum(const float* __restrict__ wf, float* __restrict__ wsumP) {
  using namespace cn;
  __shared__ float red[256];
  int o = blockIdx.x, tid = threadIdx.x;
  float s = 0.0f;
  if (o < N)
    for (int p = tid; p < N; p += 256) s += wf[(size_t)o * WF_LD + p];
  red[tid] = s; __syncthreads();
  for (int off = 128; off > 0; off >>= 1) {
    if (tid < off) red[tid] += red[tid + off];
    __syncthreads();
  }
  if (tid == 0) wsumP[o] = red[0];
}

// c2T[o][b] = je[b]·wf[o,778:] + bf[o];  A=jeh[128x512], B=wf2h[o][512]
// grid (13, 8), block 32; stores f32 [832][128]
__global__ void k_bias2(const _Float16* __restrict__ jeh, const _Float16* __restrict__ wf2h,
                        const float* __restrict__ bfP, float* __restrict__ c2T) {
  using namespace cn;
  int lane = threadIdx.x;
  int g = lane >> 4, nl = lane & 15;
  int m0 = blockIdx.y * 16;   // batch rows
  int n0b = blockIdx.x * 64;  // o cols
  const _Float16* Arow = jeh + (size_t)(m0 + nl) * DPEP;
  const _Float16* Bcol = wf2h + (size_t)(n0b + nl) * DPEP;
  v8f acc[4];
#pragma unroll
  for (int j = 0; j < 4; ++j) acc[j] = vzero8();
#pragma unroll 2
  for (int k0 = 0; k0 < DPEP; k0 += 32) {
    v16h a = loadA16(Arow + k0, g);
#pragma unroll
    for (int j = 0; j < 4; ++j) {
      v16h bt = loadBc(Bcol + (size_t)(16 * j) * DPEP + k0, g);
      acc[j] = wmma32(a, bt, acc[j]);
    }
  }
  int mb = m0 + 8 * g;
#pragma unroll
  for (int j = 0; j < 4; ++j) {
    int n = n0b + 16 * j + nl;
    float bfn = bfP[n];
    v4f lo, hi;
#pragma unroll
    for (int i = 0; i < 4; ++i) { lo[i] = acc[j][i] + bfn; hi[i] = acc[j][4 + i] + bfn; }
    *(v4f*)(c2T + (size_t)n * B + mb)     = lo;
    *(v4f*)(c2T + (size_t)n * B + mb + 4) = hi;
  }
}

// P GEMM: PmT[b][j][o] = sum_p wfh[o][p] * pf[b][p][j]
// grid (1, 49, B), block 32; A=wfh [784x800] f16; B strided from pf
__global__ void k_gemm_P(const _Float16* __restrict__ wfh, const _Float16* __restrict__ pf,
                         float* __restrict__ PmT) {
  using namespace cn;
  int lane = threadIdx.x;
  int g = lane >> 4, nl = lane & 15;
  int b = blockIdx.z;
  int m0 = blockIdx.y * 16; // o rows
  const _Float16* Arow = wfh + (size_t)(m0 + nl) * WFA_LD;
  const _Float16* pfb = pf + (size_t)b * NPP * C1;
  v8f acc[4];
#pragma unroll
  for (int j = 0; j < 4; ++j) acc[j] = vzero8();
  for (int k0 = 0; k0 < WFA_LD; k0 += 32) {
    v16h a = loadA16(Arow + k0, g);
#pragma unroll
    for (int j = 0; j < 4; ++j) {
      // B[k=p][n=j]: element pf[b][p][16j+nl], stride C1 along p (rows<832, zeros >=778)
      v16h bt = loadBs(pfb + (size_t)k0 * C1 + 16 * j + nl, C1, g);
      acc[j] = wmma32(a, bt, acc[j]);
    }
  }
  int mb = m0 + 8 * g;
  float* Pb = PmT + (size_t)b * C1 * NPP;
#pragma unroll
  for (int j = 0; j < 4; ++j) {
    int n = 16 * j + nl; // j-channel < 64
    v4f lo, hi;
#pragma unroll
    for (int i = 0; i < 4; ++i) { lo[i] = acc[j][i]; hi[i] = acc[j][4 + i]; }
    *(v4f*)(Pb + (size_t)n * NPP + mb)     = lo;
    *(v4f*)(Pb + (size_t)n * NPP + mb + 4) = hi;
  }
}

// w1 layer, B operand generated on the fly (h2 never materialized).
// K loop split at 1024 (tile boundary) => branchless, uniform generation.
// grid (13, 32, B), block 32; Y = a512 [b][n][512]
__global__ void k_gemm_w1(const _Float16* __restrict__ w1h, const float* __restrict__ gd,
                          const float* __restrict__ wsumP, const float* __restrict__ c2T,
                          const float* __restrict__ PmT,
                          const float* __restrict__ sfP, const float* __restrict__ tfP,
                          const float* __restrict__ b1, const float* __restrict__ s1,
                          const float* __restrict__ t1, _Float16* __restrict__ Y) {
  using namespace cn;
  int lane = threadIdx.x;
  int g = lane >> 4, nl = lane & 15;
  int b = blockIdx.z;
  int m0 = blockIdx.y * 16;
  int n0b = blockIdx.x * 64;
  // per-column constants (padded arrays: no guards)
  float wsn[4], sfn[4], csf[4];
  int nn[4];
#pragma unroll
  for (int j = 0; j < 4; ++j) {
    int n = n0b + 16 * j + nl;
    nn[j] = n;
    wsn[j] = wsumP[n];
    sfn[j] = sfP[n];
    csf[j] = c2T[(size_t)n * B + b] * sfn[j] + tfP[n]; // (v+c2)*sf+tf = v*sf+csf
  }
  const float* gb = gd + (size_t)b * C3;
  const float* Pb = PmT + (size_t)b * C1 * NPP;
  const _Float16* Arow = w1h + (size_t)(m0 + nl) * F;
  v8f acc[4];
#pragma unroll
  for (int j = 0; j < 4; ++j) acc[j] = vzero8();

  // ---- k in [0,1024): h2 = relu(g[b,k]*wsum[n]*sf + csf)
  for (int k0 = 0; k0 < C3; k0 += 32) {
    v16h a = loadA16(Arow + k0, g);
    float gk[16];
#pragma unroll
    for (int t = 0; t < 4; ++t) {
      v4f gv = *(const v4f*)(gb + k0 + 16 * g + 4 * t);
#pragma unroll
      for (int i = 0; i < 4; ++i) gk[4 * t + i] = gv[i];
    }
#pragma unroll
    for (int j = 0; j < 4; ++j) {
      float a1 = wsn[j] * sfn[j];
      v16h bt;
#pragma unroll
      for (int h = 0; h < 16; ++h)
        bt[h] = (_Float16)fmaxf(gk[h] * a1 + csf[j], 0.0f);
      acc[j] = wmma32(a, bt, acc[j]);
    }
  }
  // ---- k in [1024,1088): h2 = relu(P[b, n, k-1024]*sf + csf)
  for (int k0 = C3; k0 < F; k0 += 32) {
    v16h a = loadA16(Arow + k0, g);
#pragma unroll
    for (int j = 0; j < 4; ++j) {
      int n = nn[j];
      v16h bt;
#pragma unroll
      for (int h = 0; h < 16; ++h) {
        float pv = Pb[(size_t)(k0 - C3 + 16 * g + h) * NPP + n];
        bt[h] = (_Float16)fmaxf(pv * sfn[j] + csf[j], 0.0f);
      }
      acc[j] = wmma32(a, bt, acc[j]);
    }
  }
  int mb = m0 + 8 * g;
  float bi[8], sc[8], sh[8];
#pragma unroll
  for (int i = 0; i < 8; ++i) { bi[i] = b1[mb + i]; sc[i] = s1[mb + i]; sh[i] = t1[mb + i]; }
#pragma unroll
  for (int j = 0; j < 4; ++j) {
    v8h hv;
#pragma unroll
    for (int i = 0; i < 8; ++i)
      hv[i] = (_Float16)fmaxf((acc[j][i] + bi[i]) * sc[i] + sh[i], 0.0f);
    *(v8h*)(Y + ((size_t)b * NPP + nn[j]) * O1 + mb) = hv;
  }
}

// probs[b,n] = sigmoid(b4 + sum_f w4[f]*y3[b,n,f])  (y3 channel-last, contiguous)
__global__ void k_w4_sigmoid(const float* __restrict__ w4, const float* __restrict__ b4,
                             const _Float16* __restrict__ Y3, float* __restrict__ probs) {
  using namespace cn;
  __shared__ float sw[O3];
  int tid = threadIdx.x;
  for (int i = tid; i < O3; i += 256) sw[i] = w4[i];
  __syncthreads();
  int idx = blockIdx.x * 256 + tid;
  if (idx >= B * NPP) return;
  int b = idx / NPP, n = idx % NPP;
  if (n >= N) return;
  const _Float16* row = Y3 + (size_t)idx * O3;
  float s = b4[0];
#pragma unroll 2
  for (int f0 = 0; f0 < O3; f0 += 8) {
    v8h hv = *(const v8h*)(row + f0);
#pragma unroll
    for (int u = 0; u < 8; ++u) s += sw[f0 + u] * (float)hv[u];
  }
  probs[(size_t)b * N + n] = 1.0f / (1.0f + __expf(-s));
}

// Stable stream compaction: contacts-first gather of verts + counts
__global__ void k_compact(const float* __restrict__ probs, const float* __restrict__ verts,
                          float* __restrict__ contact, float* __restrict__ counts) {
  using namespace cn;
  __shared__ int sc[256];
  __shared__ int sbase;
  int b = blockIdx.x, tid = threadIdx.x;
  float* cb = contact + (size_t)b * N * 3;
  for (int i = tid; i < N * 3; i += 256) cb[i] = 0.0f;
  if (tid == 0) sbase = 0;
  __syncthreads();
  for (int start = 0; start < N; start += 256) {
    int n = start + tid;
    int m = (n < N && probs[(size_t)b * N + n] >= 0.5f) ? 1 : 0;
    sc[tid] = m;
    __syncthreads();
    for (int off = 1; off < 256; off <<= 1) {
      int v = sc[tid];
      int add = (tid >= off) ? sc[tid - off] : 0;
      __syncthreads();
      sc[tid] = v + add;
      __syncthreads();
    }
    int incl = sc[tid];
    int tot = sc[255];
    if (m) {
      int pos = sbase + incl - 1;
      const float* vp = verts + ((size_t)b * N + n) * 3;
      cb[(size_t)pos * 3 + 0] = vp[0];
      cb[(size_t)pos * 3 + 1] = vp[1];
      cb[(size_t)pos * 3 + 2] = vp[2];
    }
    __syncthreads();
    if (tid == 0) sbase += tot;
    __syncthreads();
  }
  if (tid == 0) counts[b] = (float)sbase;
}

// ---------------- launch ----------------

extern "C" void kernel_launch(void* const* d_in, const int* in_sizes, int n_in,
                              void* d_out, int out_size, void* d_ws, size_t ws_size,
                              hipStream_t stream) {
  using namespace cn;
  (void)in_sizes; (void)n_in; (void)out_size; (void)ws_size;

  // setup_inputs() dict order (params dict flattened in insertion order)
  const float* je      = (const float*)d_in[0];
  const float* verts   = (const float*)d_in[1];
  const float* normals = (const float*)d_in[2];
  const float* pw1 = (const float*)d_in[3];
  const float* pb1 = (const float*)d_in[4];
  const float* ps1 = (const float*)d_in[5];
  const float* pt1 = (const float*)d_in[6];
  const float* pw2 = (const float*)d_in[7];
  const float* pb2 = (const float*)d_in[8];
  const float* ps2 = (const float*)d_in[9];
  const float* pt2 = (const float*)d_in[10];
  const float* pw3 = (const float*)d_in[11];
  const float* pb3 = (const float*)d_in[12];
  const float* ps3 = (const float*)d_in[13];
  const float* pt3 = (const float*)d_in[14];
  const float* wf  = (const float*)d_in[15];
  const float* bf  = (const float*)d_in[16];
  const float* sf  = (const float*)d_in[17];
  const float* tf  = (const float*)d_in[18];
  const float* w1  = (const float*)d_in[19];
  const float* b1  = (const float*)d_in[20];
  const float* s1  = (const float*)d_in[21];
  const float* t1  = (const float*)d_in[22];
  const float* w2  = (const float*)d_in[23];
  const float* b2  = (const float*)d_in[24];
  const float* s2  = (const float*)d_in[25];
  const float* t2  = (const float*)d_in[26];
  const float* w3  = (const float*)d_in[27];
  const float* b3  = (const float*)d_in[28];
  const float* s3  = (const float*)d_in[29];
  const float* t3  = (const float*)d_in[30];
  const float* w4  = (const float*)d_in[31];
  const float* b4  = (const float*)d_in[32];

  float* out_probs   = (float*)d_out;
  float* out_contact = out_probs + (size_t)B * N;
  float* out_counts  = out_contact + (size_t)B * N * 3;

  uint8_t* w = (uint8_t*)d_ws;
  auto carve = [&](size_t bytes) {
    void* p = (void*)w;
    w += (bytes + 255) & ~(size_t)255;
    return p;
  };
  // packed weights (f16, zero-padded)
  _Float16* pw2h = (_Float16*)carve((size_t)C2 * C1 * 2);
  _Float16* pw3h = (_Float16*)carve((size_t)C3 * C2 * 2);
  _Float16* w1h  = (_Float16*)carve((size_t)O1 * F * 2);
  _Float16* w2h  = (_Float16*)carve((size_t)O2 * O1 * 2);
  _Float16* w3h  = (_Float16*)carve((size_t)O3 * O2 * 2);
  _Float16* jeh  = (_Float16*)carve((size_t)B * DPEP * 2);
  _Float16* wf2h = (_Float16*)carve((size_t)NPP * DPEP * 2);
  _Float16* wfh  = (_Float16*)carve((size_t)WFA_R * WFA_LD * 2);
  float* bfP   = (float*)carve((size_t)NPP * 4);
  float* sfP   = (float*)carve((size_t)NPP * 4);
  float* tfP   = (float*)carve((size_t)NPP * 4);
  float* wsumP = (float*)carve((size_t)NPP * 4);
  // activations (channel-last, rows padded to NPP)
  _Float16* pf    = (_Float16*)carve((size_t)B * NPP * C1 * 2);
  _Float16* a128  = (_Float16*)carve((size_t)B * NPP * C2 * 2);
  unsigned* gmax  = (unsigned*)carve((size_t)B * C3 * 4);
  float*    gdec  = (float*)   carve((size_t)B * C3 * 4);
  float*    c2T   = (float*)   carve((size_t)NPP * B * 4);
  float*    PmT   = (float*)   carve((size_t)B * C1 * NPP * 4);
  _Float16* a512  = (_Float16*)carve((size_t)B * NPP * O1 * 2);
  _Float16* a256  = (_Float16*)carve((size_t)B * NPP * O2 * 2);
  _Float16* a128b = (_Float16*)carve((size_t)B * NPP * O3 * 2);

  auto packf16 = [&](_Float16* dst, const float* src, int sr, int scn, int sld, int dc, int dr) {
    int total = dr * dc;
    k_pack_f16<<<(total + 255) / 256, 256, 0, stream>>>(dst, src, sr, scn, sld, dc, total);
  };
  packf16(pw2h, pw2, C2, C1, C1, C1, C2);
  packf16(pw3h, pw3, C3, C2, C2, C2, C3);
  packf16(w1h,  w1,  O1, F,  F,  F,  O1);
  packf16(w2h,  w2,  O2, O1, O1, O1, O2);
  packf16(w3h,  w3,  O3, O2, O2, O2, O3);
  packf16(jeh,  je,  B,  DPE, DPE, DPEP, B);
  packf16(wf2h, wf + N, N, DPE, WF_LD, DPEP, NPP);
  packf16(wfh,  wf,  N,  N,  WF_LD, WFA_LD, WFA_R);
  k_pack_f32<<<(NPP + 255) / 256, 256, 0, stream>>>(bfP, bf, N, NPP);
  k_pack_f32<<<(NPP + 255) / 256, 256, 0, stream>>>(sfP, sf, N, NPP);
  k_pack_f32<<<(NPP + 255) / 256, 256, 0, stream>>>(tfP, tf, N, NPP);

  k_l1<<<(B * NPP + 255) / 256, 256, 0, stream>>>(verts, normals, pw1, pb1, ps1, pt1, pf);

  k_gemm_bn_relu<<<dim3(NT64, C2 / 16, B), 32, 0, stream>>>(pw2h, C2, C1, pf,
                                                            pb2, ps2, pt2, a128);

  k_init_gmax<<<(B * C3 + 255) / 256, 256, 0, stream>>>(gmax);
  k_gemm_bn_max<<<dim3(NT64, C3 / 16, B), 32, 0, stream>>>(pw3h, a128, pb3, ps3, pt3, gmax);
  k_decode_gmax<<<(B * C3 + 255) / 256, 256, 0, stream>>>(gmax, gdec);

  k_wsum<<<NPP, 256, 0, stream>>>(wf, wsumP);
  k_bias2<<<dim3(NT64, B / 16), 32, 0, stream>>>(jeh, wf2h, bfP, c2T);
  k_gemm_P<<<dim3(1, WFA_R / 16, B), 32, 0, stream>>>(wfh, pf, PmT);

  k_gemm_w1<<<dim3(NT64, O1 / 16, B), 32, 0, stream>>>(w1h, gdec, wsumP, c2T, PmT,
                                                       sfP, tfP, b1, s1, t1, a512);

  k_gemm_bn_relu<<<dim3(NT64, O2 / 16, B), 32, 0, stream>>>(w2h, O2, O1, a512,
                                                            b2, s2, t2, a256);
  k_gemm_bn_relu<<<dim3(NT64, O3 / 16, B), 32, 0, stream>>>(w3h, O3, O2, a256,
                                                            b3, s3, t3, a128b);

  k_w4_sigmoid<<<(B * NPP + 255) / 256, 256, 0, stream>>>(w4, b4, a128b, out_probs);

  k_compact<<<B, 256, 0, stream>>>(out_probs, verts, out_contact, out_counts);
}
